// GCN_41970420417735
// MI455X (gfx1250) — matrix operand order repaired and checked by reading, not compiled, not run on hardware
//
#include <hip/hip_runtime.h>
#include <hip/hip_bf16.h>

typedef __attribute__((ext_vector_type(16))) _Float16 v16h;
typedef __attribute__((ext_vector_type(8)))  float    v8f;

// ---------------------------------------------------------------------------
// Degree / normalization
// ---------------------------------------------------------------------------
__global__ void gcn_deg_init(float* __restrict__ deg, int n) {
  int i = blockIdx.x * blockDim.x + threadIdx.x;
  if (i < n) deg[i] = 1.0f;                     // self-loop contributes 1
}

__global__ void gcn_deg_accum(const int* __restrict__ ei, float* __restrict__ deg, int E) {
  int i = blockIdx.x * blockDim.x + threadIdx.x;
  if (i < E) atomicAdd(&deg[ei[E + i]], 1.0f);  // dst row of edge_index
}

__global__ void gcn_deg_to_dinv(float* __restrict__ deg, int n) {
  int i = blockIdx.x * blockDim.x + threadIdx.x;
  if (i < n) deg[i] = rsqrtf(deg[i]);           // deg >= 1 always (self loops)
}

// ---------------------------------------------------------------------------
// WMMA GEMM: C[M,Nc] = A[M,K] @ B[K,Nc], f32 in/out, f16 matrix-core compute.
// One wave (32 threads) per 16x16 output tile. K, Nc compile-time -> fully
// unrolled K loop (K/32 back-to-back v_wmma), immediate-offset fragment loads.
// Fragment layouts per CDNA5 ISA 7.12.2 (wave32):
//   A 16x32 f16 : elem i -> K = 16*(i/8) + 2*((i/2)%4) + 8*(lane/16) + (i%2)
//   B 32x16 f16 : lane N = lane%16, elem i -> K = 16*(i/8) + 8*(lane/16) + (i%8)
//   C 16x16 f32 : elem r -> M = r + 8*(lane/16), N = lane%16
// ---------------------------------------------------------------------------
template <int K, int Nc>
__global__ void gcn_wmma_gemm(const float* __restrict__ A, const float* __restrict__ B,
                              float* __restrict__ C, int M) {
  const int row0 = blockIdx.x << 4;
  const int col0 = blockIdx.y << 4;
  const int lane = threadIdx.x;       // 0..31, full wave active (WMMA needs EXEC all 1s)
  const int half = lane >> 4;
  const int mr   = lane & 15;

  const bool full = (row0 + 16 <= M);                 // wave-uniform
  const int arow_idx = full ? (row0 + mr) : min(row0 + mr, M - 1);

  const float* __restrict__ arow = A + (size_t)arow_idx * K;
  const float* __restrict__ bcol = B + col0 + mr;

  v8f acc = {};
#pragma unroll
  for (int k0 = 0; k0 < K; k0 += 32) {
    v16h a, b;
#pragma unroll
    for (int i = 0; i < 16; ++i) {
      const int j  = i >> 1;
      const int kk = ((j >> 2) << 4) + ((j & 3) << 1) + (half << 3) + (i & 1);
      a[i] = (_Float16)arow[k0 + kk];
      b[i] = (_Float16)bcol[(k0 + ((i >> 3) << 4) + (half << 3) + (i & 7)) * Nc];
    }
    acc = __builtin_amdgcn_wmma_f32_16x16x32_f16(false, a, false, b,
                                                 (short)0, acc, false, false);
  }

  float* __restrict__ cptr = C + (size_t)(row0 + (half << 3)) * Nc + col0 + mr;
  if (full) {                                         // straight-line 8 stores
#pragma unroll
    for (int r = 0; r < 8; ++r) cptr[r * Nc] = acc[r];
  } else {                                            // ragged tail only
#pragma unroll
    for (int r = 0; r < 8; ++r)
      if (row0 + (half << 3) + r < M) cptr[r * Nc] = acc[r];
  }
}

// ---------------------------------------------------------------------------
// Aggregation: agg = D^-1/2 (A+I) D^-1/2 h        (float2-vectorized paths)
// ---------------------------------------------------------------------------
__global__ void gcn_selfloop_init(const float* __restrict__ dinv, const float* __restrict__ h,
                                  float* __restrict__ agg, long long n2, int dlogm1) {
  const long long t = (long long)blockIdx.x * blockDim.x + threadIdx.x;
  if (t >= n2) return;                               // n2 = N * D/2 float2 elements
  const int node = (int)(t >> dlogm1);
  const float w = dinv[node];
  const float ww = w * w;
  const float2 hv = ((const float2*)h)[t];
  ((float2*)agg)[t] = make_float2(ww * hv.x, ww * hv.y);  // self-loop; also zero-inits
}

template <int DLOG>   // D = 1<<DLOG features; 1<<(DLOG-1) threads per edge (float2 each)
__global__ void gcn_edge_agg(const int* __restrict__ ei, const float* __restrict__ dinv,
                             const float* __restrict__ h, float* __restrict__ agg, int E) {
  const long long t = (long long)blockIdx.x * blockDim.x + threadIdx.x;
  const int d2 = ((int)t & ((1 << (DLOG - 1)) - 1)) << 1;
  const long long e = t >> (DLOG - 1);
  if (e >= E) return;
  const int src = ei[e];
  const int dst = ei[E + e];
  const float norm = dinv[src] * dinv[dst];
  // consecutive lanes cover one h[src] row -> coalesced b64 gather per edge
  const float2 hv = *(const float2*)(h + ((size_t)src << DLOG) + d2);
  float* ap = agg + ((size_t)dst << DLOG) + d2;
  atomicAdd(ap,     norm * hv.x);
  atomicAdd(ap + 1, norm * hv.y);
}

__global__ void gcn_bias_relu(float* __restrict__ buf, const float* __restrict__ b,
                              long long n2, int dlogm1) {
  const long long t = (long long)blockIdx.x * blockDim.x + threadIdx.x;
  if (t >= n2) return;
  const int d2 = ((int)t & ((1 << dlogm1) - 1)) << 1;
  const float2 v = ((const float2*)buf)[t];
  ((float2*)buf)[t] = make_float2(fmaxf(v.x + b[d2], 0.0f),
                                  fmaxf(v.y + b[d2 + 1], 0.0f));
}

// ---------------------------------------------------------------------------
// Final: out[i,:] = log_softmax(agg2[idx[i],:] + b2); one wave32 per row (32 classes)
// ---------------------------------------------------------------------------
__global__ void gcn_logsoftmax_gather(const float* __restrict__ agg2, const float* __restrict__ b2,
                                      const int* __restrict__ idx, float* __restrict__ out,
                                      int nIdx) {
  const int lane = threadIdx.x & 31;
  const int row  = (blockIdx.x * blockDim.x + threadIdx.x) >> 5;
  if (row >= nIdx) return;                       // whole-wave uniform exit
  const int node = idx[row];
  const float v = agg2[(size_t)node * 32 + lane] + b2[lane];
  float m = v;
#pragma unroll
  for (int off = 16; off > 0; off >>= 1) m = fmaxf(m, __shfl_xor(m, off, 32));
  float s = expf(v - m);
#pragma unroll
  for (int off = 16; off > 0; off >>= 1) s += __shfl_xor(s, off, 32);
  out[(size_t)row * 32 + lane] = v - m - logf(s);
}

// ---------------------------------------------------------------------------
extern "C" void kernel_launch(void* const* d_in, const int* in_sizes, int n_in,
                              void* d_out, int out_size, void* d_ws, size_t ws_size,
                              hipStream_t stream) {
  const int*   idx = (const int*)  d_in[0];
  const float* x   = (const float*)d_in[1];
  const int*   ei  = (const int*)  d_in[2];
  const float* W1  = (const float*)d_in[3];
  const float* b1  = (const float*)d_in[4];
  const float* W2  = (const float*)d_in[5];
  const float* b2  = (const float*)d_in[6];
  float* out = (float*)d_out;

  const int D_IN = 128, D_HID = 64, D_OUT = 32;
  const int N    = in_sizes[1] / D_IN;   // 100000
  const int E    = in_sizes[2] / 2;      // 1600000
  const int NIDX = in_sizes[0];          // 1024

  // Workspace layout (floats, offsets rounded to float4 for vector alignment):
  //   dinv[Na] | hpre[64*Na] | agg1/h1[64*Na]
  // hpre region reused after layer-1 aggregation: h2pre[32*Na] + agg2[32*Na].
  const size_t Na = ((size_t)N + 3) & ~(size_t)3;
  float* ws    = (float*)d_ws;
  float* dinv  = ws;
  float* hpre  = ws + Na;                        // gemm1 output (N x 64)
  float* h1    = ws + 65 * Na;                   // agg1 -> (bias,relu) -> h1 (N x 64)
  float* h2pre = hpre;                           // gemm2 output (N x 32)
  float* agg2  = hpre + 32 * Na;                 // layer-2 aggregation (N x 32)

  const int TB = 256;
  // 1) degrees -> dinv
  gcn_deg_init   <<<(N + TB - 1) / TB, TB, 0, stream>>>(dinv, N);
  gcn_deg_accum  <<<(E + TB - 1) / TB, TB, 0, stream>>>(ei, dinv, E);
  gcn_deg_to_dinv<<<(N + TB - 1) / TB, TB, 0, stream>>>(dinv, N);

  // 2) layer 1: h = x @ W1 (WMMA), aggregate, bias+relu
  {
    dim3 g((N + 15) / 16, D_HID / 16);
    gcn_wmma_gemm<128, 64><<<g, 32, 0, stream>>>(x, W1, hpre, N);
  }
  {
    const long long n2 = (long long)N * (D_HID / 2);
    gcn_selfloop_init<<<(unsigned)((n2 + TB - 1) / TB), TB, 0, stream>>>(dinv, hpre, h1, n2, 5);
    const long long te = (long long)E * (D_HID / 2);
    gcn_edge_agg<6><<<(unsigned)((te + TB - 1) / TB), TB, 0, stream>>>(ei, dinv, hpre, h1, E);
    gcn_bias_relu<<<(unsigned)((n2 + TB - 1) / TB), TB, 0, stream>>>(h1, b1, n2, 5);
  }

  // 3) layer 2: h2 = h1 @ W2 (WMMA), aggregate
  {
    dim3 g((N + 15) / 16, D_OUT / 16);
    gcn_wmma_gemm<64, 32><<<g, 32, 0, stream>>>(h1, W2, h2pre, N);
  }
  {
    const long long n2 = (long long)N * (D_OUT / 2);
    gcn_selfloop_init<<<(unsigned)((n2 + TB - 1) / TB), TB, 0, stream>>>(dinv, h2pre, agg2, n2, 4);
    const long long te = (long long)E * (D_OUT / 2);
    gcn_edge_agg<5><<<(unsigned)((te + TB - 1) / TB), TB, 0, stream>>>(ei, dinv, h2pre, agg2, E);
  }

  // 4) bias + log_softmax + gather selected rows
  gcn_logsoftmax_gather<<<(NIDX * 32 + TB - 1) / TB, TB, 0, stream>>>(agg2, b2, idx, out, NIDX);
}
